// DHGCN_sceneseg_53893249630266
// MI455X (gfx1250) — matrix operand order repaired and checked by reading, not compiled, not run on hardware
//
#include <hip/hip_runtime.h>

// ---------------------------------------------------------------------------
// DHGCN scene segmentation forward for MI455X (gfx1250, wave32, WMMA).
// All GEMMs go through v_wmma_f32_16x16x32_f16 (f32->f16 staging, f32 acc).
// Block tile 32x128, 8 waves x (16x32) strips -> 2 WMMA per wave per K-step.
// graph_feature gather is fused into the edge-GEMM B staging.
// Ragged tiles use clamped-address unconditional loads + select (no exec
// predication, no per-element load drains).
// ---------------------------------------------------------------------------

typedef __attribute__((ext_vector_type(16))) _Float16 v16h;
typedef __attribute__((ext_vector_type(8)))  _Float16 v8h;
typedef __attribute__((ext_vector_type(4)))  _Float16 v4h;
typedef __attribute__((ext_vector_type(8)))  float    v8f;

constexpr int Bz  = 4;
constexpr int Nn  = 4096;
constexpr int Pp  = 27;
constexpr int Kn  = 20;
constexpr int Mpt = Bz * Nn;        // 16384 points total
constexpr int Mk  = Mpt * Kn;       // 327680 point*k
constexpr int Mpp = Bz * Pp * Pp;   // 2916 edge positions
constexpr int Mp  = Bz * Pp;        // 108 voxels total

// ---------------------------------------------------------------------------
// Generic WMMA GEMM: Y[Cout x M] = W[Cout x Cin] * X[Cin x M]   (f32 in/out)
// ---------------------------------------------------------------------------
__global__ void gemm_wmma(const float* __restrict__ W, const float* __restrict__ X,
                          float* __restrict__ Y, int Cout, int Cin, int M) {
  __shared__ _Float16 As[32][40];    // row stride 80B  (16B multiple)
  __shared__ _Float16 Bs[32][136];   // row stride 272B (16B multiple)
  const int m0 = blockIdx.x * 128, o0 = blockIdx.y * 32;
  const int tid = threadIdx.x, lane = tid & 31, wave = tid >> 5;
  const int wm = wave >> 2, wn = wave & 3;
  const bool fullM = (m0 + 128 <= M), fullO = (o0 + 32 <= Cout);
  v8f acc0 = {}, acc1 = {};
  const int Kpad = (Cin + 31) & ~31;
  for (int k0 = 0; k0 < Kpad; k0 += 32) {
    const bool fullK = (k0 + 32 <= Cin);
    // ---- stage A: 32x32 (4 contiguous per thread) ----
    {
      const int ar = tid >> 3, ac = (tid & 7) * 4;
      v4h d;
      if (fullO && fullK) {
        const float* src = W + (size_t)(o0 + ar) * Cin + k0 + ac;
        d[0] = (_Float16)src[0]; d[1] = (_Float16)src[1];
        d[2] = (_Float16)src[2]; d[3] = (_Float16)src[3];
      } else {
        const int o = o0 + ar, oc = min(o, Cout - 1);
#pragma unroll
        for (int j = 0; j < 4; ++j) {
          int c = k0 + ac + j, cc = min(c, Cin - 1);
          float v = W[(size_t)oc * Cin + cc];           // unconditional, clamped
          d[j] = (o < Cout && c < Cin) ? (_Float16)v : (_Float16)0.f;
        }
      }
      *(v4h*)&As[ar][ac] = d;
    }
    // ---- stage B: 32x128 (16 contiguous per thread) ----
    {
      const int br = tid >> 3, bc = (tid & 7) * 16;
      v8h d0, d1;
      if (fullM && fullK) {
        const float* src = X + (size_t)(k0 + br) * M + m0 + bc;
        float4 f0 = *(const float4*)src;
        float4 f1 = *(const float4*)(src + 4);
        float4 f2 = *(const float4*)(src + 8);
        float4 f3 = *(const float4*)(src + 12);
        d0[0] = (_Float16)f0.x; d0[1] = (_Float16)f0.y;
        d0[2] = (_Float16)f0.z; d0[3] = (_Float16)f0.w;
        d0[4] = (_Float16)f1.x; d0[5] = (_Float16)f1.y;
        d0[6] = (_Float16)f1.z; d0[7] = (_Float16)f1.w;
        d1[0] = (_Float16)f2.x; d1[1] = (_Float16)f2.y;
        d1[2] = (_Float16)f2.z; d1[3] = (_Float16)f2.w;
        d1[4] = (_Float16)f3.x; d1[5] = (_Float16)f3.y;
        d1[6] = (_Float16)f3.z; d1[7] = (_Float16)f3.w;
      } else {
        const int c = k0 + br, cc = min(c, Cin - 1);
#pragma unroll
        for (int j = 0; j < 8; ++j) {
          int m = m0 + bc + j, mm = min(m, M - 1);
          float v = X[(size_t)cc * M + mm];
          d0[j] = (c < Cin && m < M) ? (_Float16)v : (_Float16)0.f;
        }
#pragma unroll
        for (int j = 0; j < 8; ++j) {
          int m = m0 + bc + 8 + j, mm = min(m, M - 1);
          float v = X[(size_t)cc * M + mm];
          d1[j] = (c < Cin && m < M) ? (_Float16)v : (_Float16)0.f;
        }
      }
      *(v8h*)&Bs[br][bc] = d0;
      *(v8h*)&Bs[br][bc + 8] = d1;
    }
    __syncthreads();
    v16h a, b0, b1;
    const int h = lane >> 4, rA = wm * 16 + (lane & 15);
#pragma unroll
    for (int e = 0; e < 16; ++e) {
      int v2 = e >> 1;
      int kk = (v2 < 4) ? (h * 8 + (v2 << 1) + (e & 1))
                        : (16 + h * 8 + ((v2 - 4) << 1) + (e & 1));
      a[e]  = As[rA][kk];
      b0[e] = Bs[kk][wn * 32 + (lane & 15)];
      b1[e] = Bs[kk][wn * 32 + 16 + (lane & 15)];
    }
    acc0 = __builtin_amdgcn_wmma_f32_16x16x32_f16(false, a, false, b0,
                                                  (short)0, acc0, false, false);
    acc1 = __builtin_amdgcn_wmma_f32_16x16x32_f16(false, a, false, b1,
                                                  (short)0, acc1, false, false);
    __syncthreads();
  }
#pragma unroll
  for (int r = 0; r < 8; ++r) {
    int row = o0 + wm * 16 + (lane >> 4) * 8 + r;
    int col0 = m0 + wn * 32 + (lane & 15);
    if (row < Cout) {
      if (col0 < M)      Y[(size_t)row * M + col0]      = acc0[r];
      if (col0 + 16 < M) Y[(size_t)row * M + col0 + 16] = acc1[r];
    }
  }
}

// ---------------------------------------------------------------------------
// WMMA GEMM with fused graph_feature gather. Cin = 2*Cfeat:
//   c <  Cfeat : x[c, nbr] - x[c, ctr]     c >= Cfeat : x[c-Cfeat, ctr]
// m = (b*N + n)*Kn + kk ; Xc is (Cfeat, B*N) channel-major; idx is (B*N, Kn).
// 128 columns per block: 2 threads per column, 16 k-rows each; the idx load
// and bn/kk decode hoist out of the K loop.
// ---------------------------------------------------------------------------
__global__ void gemm_wmma_edge(const float* __restrict__ W, const float* __restrict__ Xc,
                               const int* __restrict__ idx, float* __restrict__ Y,
                               int Cout, int Cfeat, int M) {
  __shared__ _Float16 As[32][40];
  __shared__ _Float16 Bs[32][136];
  const int Cin = 2 * Cfeat;
  const int m0 = blockIdx.x * 128, o0 = blockIdx.y * 32;
  const int tid = threadIdx.x, lane = tid & 31, wave = tid >> 5;
  const int wm = wave >> 2, wn = wave & 3;
  const bool fullO = (o0 + 32 <= Cout);
  // per-thread column decode (hoisted out of the K loop)
  const int bc = tid & 127, half = tid >> 7;       // rows br = r*2 + half
  const int m = m0 + bc;
  const int mc = min(m, M - 1);
  const int bnc = mc / Kn, kkc = mc - bnc * Kn;
  const int bb = bnc >> 12;                        // N = 4096
  const int nb = idx[bnc * Kn + kkc];
  const float* ctrp = Xc + bnc;
  const float* nbrp = Xc + (bb << 12) + nb;
  v8f acc0 = {}, acc1 = {};
  const int Kpad = (Cin + 31) & ~31;
  for (int k0 = 0; k0 < Kpad; k0 += 32) {
    const bool fullK = (k0 + 32 <= Cin);
    // ---- stage A ----
    {
      const int ar = tid >> 3, ac = (tid & 7) * 4;
      v4h d;
      if (fullO && fullK) {
        const float* src = W + (size_t)(o0 + ar) * Cin + k0 + ac;
        d[0] = (_Float16)src[0]; d[1] = (_Float16)src[1];
        d[2] = (_Float16)src[2]; d[3] = (_Float16)src[3];
      } else {
        const int o = o0 + ar, oc = min(o, Cout - 1);
#pragma unroll
        for (int j = 0; j < 4; ++j) {
          int c = k0 + ac + j, cc = min(c, Cin - 1);
          float v = W[(size_t)oc * Cin + cc];
          d[j] = (o < Cout && c < Cin) ? (_Float16)v : (_Float16)0.f;
        }
      }
      *(v4h*)&As[ar][ac] = d;
    }
    // ---- stage B (fused gather, column-per-thread, 16 k-rows) ----
    if (fullK && m < M) {
#pragma unroll
      for (int r = 0; r < 16; ++r) {
        int br = r * 2 + half;
        int c = k0 + br;
        float v = (c < Cfeat) ? (nbrp[(size_t)c * Mpt] - ctrp[(size_t)c * Mpt])
                              : ctrp[(size_t)(c - Cfeat) * Mpt];
        Bs[br][bc] = (_Float16)v;
      }
    } else {
#pragma unroll
      for (int r = 0; r < 16; ++r) {
        int br = r * 2 + half;
        int c = k0 + br, cc = min(c, Cin - 1);
        float v = (cc < Cfeat) ? (nbrp[(size_t)cc * Mpt] - ctrp[(size_t)cc * Mpt])
                               : ctrp[(size_t)(cc - Cfeat) * Mpt];
        Bs[br][bc] = (c < Cin && m < M) ? (_Float16)v : (_Float16)0.f;
      }
    }
    __syncthreads();
    v16h a, b0, b1;
    const int h = lane >> 4, rA = wm * 16 + (lane & 15);
#pragma unroll
    for (int e = 0; e < 16; ++e) {
      int v2 = e >> 1;
      int kk = (v2 < 4) ? (h * 8 + (v2 << 1) + (e & 1))
                        : (16 + h * 8 + ((v2 - 4) << 1) + (e & 1));
      a[e]  = As[rA][kk];
      b0[e] = Bs[kk][wn * 32 + (lane & 15)];
      b1[e] = Bs[kk][wn * 32 + 16 + (lane & 15)];
    }
    acc0 = __builtin_amdgcn_wmma_f32_16x16x32_f16(false, a, false, b0,
                                                  (short)0, acc0, false, false);
    acc1 = __builtin_amdgcn_wmma_f32_16x16x32_f16(false, a, false, b1,
                                                  (short)0, acc1, false, false);
    __syncthreads();
  }
#pragma unroll
  for (int r = 0; r < 8; ++r) {
    int row = o0 + wm * 16 + (lane >> 4) * 8 + r;
    int col0 = m0 + wn * 32 + (lane & 15);
    if (row < Cout) {
      if (col0 < M)      Y[(size_t)row * M + col0]      = acc0[r];
      if (col0 + 16 < M) Y[(size_t)row * M + col0 + 16] = acc1[r];
    }
  }
}

// ---------------------------------------------------------------------------
// BatchNorm (training stats) + LeakyReLU(0.2)
// ---------------------------------------------------------------------------
__global__ void bn_stats(const float* __restrict__ Y, float* __restrict__ st, int M) {
  const int ch = blockIdx.x;
  const float* p = Y + (size_t)ch * M;
  float s = 0.f, q = 0.f;
  for (int i = threadIdx.x; i < M; i += 256) { float v = p[i]; s += v; q += v * v; }
  __shared__ float ss[256], qq[256];
  ss[threadIdx.x] = s; qq[threadIdx.x] = q; __syncthreads();
  for (int st2 = 128; st2 > 0; st2 >>= 1) {
    if (threadIdx.x < st2) { ss[threadIdx.x] += ss[threadIdx.x + st2];
                             qq[threadIdx.x] += qq[threadIdx.x + st2]; }
    __syncthreads();
  }
  if (threadIdx.x == 0) {
    float m = ss[0] / M;
    float var = qq[0] / M - m * m;
    st[2 * ch] = m;
    st[2 * ch + 1] = rsqrtf(fmaxf(var, 0.f) + 1e-5f);
  }
}

__global__ void bn_apply_lrelu(float* __restrict__ Y, const float* __restrict__ st,
                               const float* __restrict__ g, const float* __restrict__ b,
                               int M, long total) {
  long i = (long)blockIdx.x * blockDim.x + threadIdx.x;
  if (i >= total) return;
  int c = (int)(i / M);
  float y = (Y[i] - st[2 * c]) * st[2 * c + 1] * g[c] + b[c];
  Y[i] = y > 0.f ? y : 0.2f * y;
}

// ---------------------------------------------------------------------------
// Small utility kernels
// ---------------------------------------------------------------------------
__global__ void pack_x(const float* __restrict__ x, float* __restrict__ xc) {
  int i = blockIdx.x * blockDim.x + threadIdx.x;
  if (i >= 9 * Mpt) return;
  int c = i / Mpt, bn = i % Mpt, b = bn / Nn, n = bn % Nn;
  xc[i] = x[(b * 9 + c) * Nn + n];
}

__global__ void mask_k(const int* __restrict__ p2v, const int* __restrict__ prand,
                       float* __restrict__ mask) {
  int i = blockIdx.x * blockDim.x + threadIdx.x;
  if (i >= Mpt * Pp) return;
  int bn = i / Pp, p = i % Pp, b = bn / Nn;
  mask[i] = (p2v[bn] == prand[b * Pp + p]) ? 1.f : 0.f;
}

__global__ void cnt_k(const float* __restrict__ mask, float* __restrict__ cnt) {
  int bp = blockIdx.x, b = bp / Pp, p = bp % Pp;
  float s = 0.f;
  for (int n = threadIdx.x; n < Nn; n += 256) s += mask[(size_t)(b * Nn + n) * Pp + p];
  __shared__ float ss[256];
  ss[threadIdx.x] = s; __syncthreads();
  for (int st = 128; st > 0; st >>= 1) {
    if (threadIdx.x < st) ss[threadIdx.x] += ss[threadIdx.x + st];
    __syncthreads();
  }
  if (threadIdx.x == 0) cnt[bp] = fmaxf(ss[0], 1.f);
}

__global__ void part_pool_k(const float* __restrict__ X, const float* __restrict__ mask,
                            const float* __restrict__ cnt, float* __restrict__ out) {
  int id = blockIdx.x;                       // grid = C * Mp
  int c = id / Mp, bp = id % Mp, b = bp / Pp, p = bp % Pp;
  float s = 0.f;
  for (int n = threadIdx.x; n < Nn; n += 128)
    s += X[(size_t)c * Mpt + b * Nn + n] * mask[(size_t)(b * Nn + n) * Pp + p];
  __shared__ float ss[128];
  ss[threadIdx.x] = s; __syncthreads();
  for (int st = 64; st > 0; st >>= 1) {
    if (threadIdx.x < st) ss[threadIdx.x] += ss[threadIdx.x + st];
    __syncthreads();
  }
  if (threadIdx.x == 0) out[c * Mp + bp] = ss[0] / cnt[bp];
}

__global__ void add_k(const float* __restrict__ a, const float* __restrict__ b,
                      float* __restrict__ o, int total) {
  int i = blockIdx.x * blockDim.x + threadIdx.x;
  if (i < total) o[i] = a[i] + b[i];
}

__global__ void norms_k(const float* __restrict__ Xc, int c0, int Cc,
                        float* __restrict__ xx) {
  int bn = blockIdx.x * blockDim.x + threadIdx.x;
  if (bn >= Mpt) return;
  float s = 0.f;
  for (int c = 0; c < Cc; ++c) { float v = Xc[(size_t)(c0 + c) * Mpt + bn]; s += v * v; }
  xx[bn] = s;
}

// KNN: one block per point, 128 threads. Neighbor features staged coalesced
// into LDS per 128-wide tile; register top-20 insertion + LDS log-merge.
__global__ void knn_k(const float* __restrict__ Xc, const float* __restrict__ xx,
                      int c0, int Cc, int* __restrict__ idxo) {
  const int bn = blockIdx.x, b = bn / Nn;
  __shared__ float ctr[64];
  __shared__ float tile[64 * 128];           // [c][m_local]
  __shared__ float cv[128 * Kn];
  __shared__ int   ci[128 * Kn];
  for (int c = threadIdx.x; c < Cc; c += 128) ctr[c] = Xc[(size_t)(c0 + c) * Mpt + bn];
  const float xxn = xx[bn];
  float tv[Kn]; int ti[Kn];
#pragma unroll
  for (int j = 0; j < Kn; ++j) { tv[j] = -3.4e38f; ti[j] = 0; }
  for (int t = 0; t < Nn; t += 128) {
    __syncthreads();
    for (int c = 0; c < Cc; ++c)             // coalesced across threads per channel
      tile[c * 128 + threadIdx.x] = Xc[(size_t)(c0 + c) * Mpt + b * Nn + t + threadIdx.x];
    __syncthreads();
    const int m = t + threadIdx.x;
    float dot = 0.f;
    for (int c = 0; c < Cc; ++c) dot += ctr[c] * tile[c * 128 + threadIdx.x];
    float d = 2.f * dot - xxn - xx[b * Nn + m];   // negative squared distance
    if (d > tv[Kn - 1]) {
      int j = Kn - 1;
      while (j > 0 && tv[j - 1] < d) { tv[j] = tv[j - 1]; ti[j] = ti[j - 1]; --j; }
      tv[j] = d; ti[j] = m;
    }
  }
  for (int j = 0; j < Kn; ++j) { cv[threadIdx.x * Kn + j] = tv[j]; ci[threadIdx.x * Kn + j] = ti[j]; }
  __syncthreads();
  for (int stride = 64; stride > 0; stride >>= 1) {
    if (threadIdx.x < stride) {
      float* A  = &cv[threadIdx.x * Kn];            int* Ai = &ci[threadIdx.x * Kn];
      float* Bv = &cv[(threadIdx.x + stride) * Kn]; int* Bi = &ci[(threadIdx.x + stride) * Kn];
      float ov[Kn]; int oi[Kn]; int ia = 0, ib = 0;
#pragma unroll
      for (int j = 0; j < Kn; ++j) {
        if (A[ia] >= Bv[ib]) { ov[j] = A[ia]; oi[j] = Ai[ia]; ++ia; }
        else                 { ov[j] = Bv[ib]; oi[j] = Bi[ib]; ++ib; }
      }
      for (int j = 0; j < Kn; ++j) { A[j] = ov[j]; Ai[j] = oi[j]; }
    }
    __syncthreads();
  }
  if (threadIdx.x < Kn) idxo[bn * Kn + threadIdx.x] = ci[threadIdx.x];
}

__global__ void maxk_k(const float* __restrict__ Y, float* __restrict__ X) {
  int i = blockIdx.x * blockDim.x + threadIdx.x;
  if (i >= 64 * Mpt) return;
  int c = i / Mpt, bn = i % Mpt;
  const float* p = Y + (size_t)c * Mk + (size_t)bn * Kn;
  float mx = p[0];
  for (int k = 1; k < Kn; ++k) mx = fmaxf(mx, p[k]);
  X[i] = mx;
}

// hop-GCN helpers --------------------------------------------------------
__global__ void edge_feat(const float* __restrict__ pf, float* __restrict__ ef) {
  int i = blockIdx.x * blockDim.x + threadIdx.x;
  if (i >= 128 * Mpp) return;
  int c = i / Mpp, m = i % Mpp;
  int b = m / (Pp * Pp), r = m % (Pp * Pp), ii = r / Pp, jj = r % Pp;
  ef[i] = (c < 64) ? (pf[c * Mp + b * Pp + jj] - pf[c * Mp + b * Pp + ii])
                   : pf[(c - 64) * Mp + b * Pp + ii];
}

__global__ void att_score(const float* __restrict__ EF, const float* __restrict__ w,
                          const float* __restrict__ ghop, float* __restrict__ s) {
  int i = blockIdx.x * blockDim.x + threadIdx.x;
  if (i >= Bz * 4 * Pp * Pp) return;
  int bh = i / (Pp * Pp), r = i % (Pp * Pp);
  int b = bh >> 2, h = bh & 3;
  int m = b * Pp * Pp + r;
  float sc = ghop ? ghop[m] : 1.f;
  float acc = 0.f;
  for (int t = 0; t < 16; ++t) {
    float v = sc * EF[(size_t)(t * 4 + h) * Mpp + m];
    acc += w[t] * (v > 0.f ? v : 0.2f * v);
  }
  s[i] = acc;
}

__global__ void softmax27(float* __restrict__ s) {
  int row = blockIdx.x * blockDim.x + threadIdx.x;
  if (row >= Bz * 4 * Pp) return;
  float* p = s + row * Pp;
  float mx = p[0];
  for (int j = 1; j < Pp; ++j) mx = fmaxf(mx, p[j]);
  float sm = 0.f;
  for (int j = 0; j < Pp; ++j) { p[j] = __expf(p[j] - mx); sm += p[j]; }
  float inv = 1.f / sm;
  for (int j = 0; j < Pp; ++j) p[j] *= inv;
}

__global__ void att_scale(const float* __restrict__ EF, const float* __restrict__ a,
                          float* __restrict__ out) {
  int i = blockIdx.x * blockDim.x + threadIdx.x;
  if (i >= 64 * Mpp) return;
  int c = i / Mpp, m = i % Mpp;
  int b = m / (Pp * Pp), r = m % (Pp * Pp);
  out[i] = EF[i] * a[(b * 4 + (c & 3)) * (Pp * Pp) + r];
}

__global__ void hop_ghop(const float* __restrict__ lg, float* __restrict__ gh) {
  int m = blockIdx.x * blockDim.x + threadIdx.x;
  if (m >= Mpp) return;
  float best = lg[m]; int bi = 0;
  for (int c = 1; c < 5; ++c) { float v = lg[(size_t)c * Mpp + m]; if (v > best) { best = v; bi = c; } }
  float hp = (float)bi;
  gh[m] = __expf(-hp * hp * 0.5f) * 0.3989422804014327f;
}

__global__ void att_gmean(const float* __restrict__ a, const float* __restrict__ EF,
                          float* __restrict__ g) {
  int i = blockIdx.x * blockDim.x + threadIdx.x;
  if (i >= 64 * Mp) return;
  int c = i / Mp, bp = i % Mp, b = bp / Pp, ii = bp % Pp;
  float s = 0.f;
  for (int j = 0; j < Pp; ++j)
    s += a[(b * 4 + (c & 3)) * (Pp * Pp) + ii * Pp + j] *
         EF[(size_t)c * Mpp + b * Pp * Pp + ii * Pp + j];
  g[i] = s * (1.f / Pp);
}

__global__ void p2p_add(const float* __restrict__ g, const float* __restrict__ mask,
                        float* __restrict__ X) {
  int i = blockIdx.x * blockDim.x + threadIdx.x;
  if (i >= 64 * Mpt) return;
  int c = i / Mpt, bn = i % Mpt, b = bn / Nn;
  float s = 0.f;
  for (int p = 0; p < Pp; ++p) s += g[c * Mp + b * Pp + p] * mask[(size_t)bn * Pp + p];
  X[i] += s;
}

// final head -------------------------------------------------------------
__global__ void concat3(const float* __restrict__ x1, const float* __restrict__ x2,
                        const float* __restrict__ x3, float* __restrict__ o) {
  int i = blockIdx.x * blockDim.x + threadIdx.x;
  if (i >= 192 * Mpt) return;
  int c = i / Mpt, bn = i % Mpt;
  o[i] = (c < 64) ? x1[(size_t)c * Mpt + bn]
       : (c < 128) ? x2[(size_t)(c - 64) * Mpt + bn]
                   : x3[(size_t)(c - 128) * Mpt + bn];
}

__global__ void gmax_k(const float* __restrict__ e, float* __restrict__ gm) {
  int b = blockIdx.x >> 10, c = blockIdx.x & 1023;
  __shared__ float ss[256];
  float mx = -3.4e38f;
  for (int n = threadIdx.x; n < Nn; n += 256)
    mx = fmaxf(mx, e[(size_t)c * Mpt + b * Nn + n]);
  ss[threadIdx.x] = mx; __syncthreads();
  for (int st = 128; st > 0; st >>= 1) {
    if (threadIdx.x < st) ss[threadIdx.x] = fmaxf(ss[threadIdx.x], ss[threadIdx.x + st]);
    __syncthreads();
  }
  if (threadIdx.x == 0) gm[blockIdx.x] = ss[0];
}

__global__ void concat2(const float* __restrict__ gm, const float* __restrict__ x1,
                        const float* __restrict__ x2, const float* __restrict__ x3,
                        float* __restrict__ o) {
  long i = (long)blockIdx.x * blockDim.x + threadIdx.x;
  if (i >= (long)1216 * Mpt) return;
  int c = (int)(i / Mpt), bn = (int)(i % Mpt), b = bn / Nn;
  float v;
  if (c < 1024)       v = gm[b * 1024 + c];
  else if (c < 1088)  v = x1[(size_t)(c - 1024) * Mpt + bn];
  else if (c < 1152)  v = x2[(size_t)(c - 1088) * Mpt + bn];
  else                v = x3[(size_t)(c - 1152) * Mpt + bn];
  o[i] = v;
}

__global__ void write_o(const float* __restrict__ o9, float* __restrict__ out) {
  int i = blockIdx.x * blockDim.x + threadIdx.x;
  if (i >= 13 * Mpt) return;
  int c = i / Mpt, bn = i % Mpt, b = bn / Nn, n = bn % Nn;
  out[(b * 13 + c) * Nn + n] = o9[i];
}

__global__ void write_h(const float* __restrict__ lg, float* __restrict__ out) {
  int i = blockIdx.x * blockDim.x + threadIdx.x;
  if (i >= 5 * Mpp) return;
  int c = i / Mpp, m = i % Mpp, b = m / (Pp * Pp), r = m % (Pp * Pp);
  out[(b * 5 + c) * (Pp * Pp) + r] = lg[i];
}

// ---------------------------------------------------------------------------
// Host orchestration
// ---------------------------------------------------------------------------
extern "C" void kernel_launch(void* const* d_in, const int* in_sizes, int n_in,
                              void* d_out, int out_size, void* d_ws, size_t ws_size,
                              hipStream_t stream) {
  (void)in_sizes; (void)n_in; (void)out_size; (void)ws_size;
  const float* x    = (const float*)d_in[0];
  const int* p2v    = (const int*)d_in[1];
  const int* prand  = (const int*)d_in[2];
  auto F = [&](int i) { return (const float*)d_in[i]; };
  // param indices (flattened in dict-insertion order)
  // 3..5 pce, 6.. c1..c8 (w,g,b), 30 c9_w, 31/45/59 hop1/2/3 bases.

  // ---- workspace bump allocator (floats) ----
  float* wsp = (float*)d_ws; size_t off = 0;
  auto alloc = [&](size_t n) { float* p = wsp + off; off += n; return p; };
  float* xc    = alloc((size_t)9 * Mpt);
  float* mask  = alloc((size_t)Mpt * Pp);
  float* cnt   = alloc(Mp);
  float* xx    = alloc(Mpt);
  int*   idx   = (int*)alloc((size_t)Mpt * Kn);
  float* pp    = alloc((size_t)64 * Mp);
  float* pce   = alloc((size_t)64 * Mp);
  float* pf    = alloc((size_t)64 * Mp);
  float* gbuf  = alloc((size_t)64 * Mp);
  float* x1    = alloc((size_t)64 * Mpt);
  float* x2    = alloc((size_t)64 * Mpt);
  float* x3    = alloc((size_t)64 * Mpt);
  float* ef128 = alloc((size_t)128 * Mpp);
  float* ef1   = alloc((size_t)64 * Mpp);
  float* efs   = alloc((size_t)64 * Mpp);
  float* ef2   = alloc((size_t)64 * Mpp);
  float* hb    = alloc((size_t)32 * Mpp);
  float* lgts  = alloc((size_t)5 * Mpp);
  float* att   = alloc((size_t)Bz * 4 * Pp * Pp);
  float* ghop  = alloc(Mpp);
  float* gm    = alloc((size_t)Bz * 1024);
  float* cat192= alloc((size_t)192 * Mpt);
  float* o7    = alloc((size_t)512 * Mpt);
  float* o8    = alloc((size_t)256 * Mpt);
  float* o9    = alloc((size_t)13 * Mpt);
  float* stats = alloc(2048);
  float* YA    = alloc((size_t)64 * Mk);   // also holds c6 output (1024 x Mpt)
  float* YB    = alloc((size_t)64 * Mk);   // also holds 1216 x Mpt concat

  auto blocks = [](long total) { return (unsigned)((total + 255) / 256); };

  // GEMM + BN(train) + LeakyReLU
  auto cbl = [&](const float* W, const float* Xin, float* Y,
                 int Cout, int Cin, int M, const float* g, const float* bb) {
    dim3 grid((M + 127) / 128, (Cout + 31) / 32);
    gemm_wmma<<<grid, 256, 0, stream>>>(W, Xin, Y, Cout, Cin, M);
    bn_stats<<<Cout, 256, 0, stream>>>(Y, stats, M);
    long tot = (long)Cout * M;
    bn_apply_lrelu<<<blocks(tot), 256, 0, stream>>>(Y, stats, g, bb, M, tot);
  };
  auto cble = [&](const float* W, const float* xs, float* Y,
                  int Cout, int Cfeat, const float* g, const float* bb) {
    dim3 grid((Mk + 127) / 128, (Cout + 31) / 32);
    gemm_wmma_edge<<<grid, 256, 0, stream>>>(W, xs, idx, Y, Cout, Cfeat, Mk);
    bn_stats<<<Cout, 256, 0, stream>>>(Y, stats, Mk);
    long tot = (long)Cout * Mk;
    bn_apply_lrelu<<<blocks(tot), 256, 0, stream>>>(Y, stats, g, bb, Mk, tot);
  };

  // ---- preliminaries ----
  pack_x<<<blocks(9 * Mpt), 256, 0, stream>>>(x, xc);
  mask_k<<<blocks((long)Mpt * Pp), 256, 0, stream>>>(p2v, prand, mask);
  cnt_k<<<Mp, 256, 0, stream>>>(mask, cnt);

  // pce = cbl1(part_pool(x[:,6:], mask))
  part_pool_k<<<3 * Mp, 128, 0, stream>>>(xc + (size_t)6 * Mpt, mask, cnt, pp);
  cbl(F(3), pp, pce, 64, 3, Mp, F(4), F(5));

  // hop-GCN block
  auto hop = [&](int pb, float* xi, int hout) {
    part_pool_k<<<64 * Mp, 128, 0, stream>>>(xi, mask, cnt, pp);
    add_k<<<blocks(64 * Mp), 256, 0, stream>>>(pce, pp, pf, 64 * Mp);
    edge_feat<<<blocks((long)128 * Mpp), 256, 0, stream>>>(pf, ef128);
    cbl(F(pb + 0), ef128, ef1, 64, 128, Mpp, F(pb + 1), F(pb + 2));   // ec1
    att_score<<<blocks(Bz * 4 * Pp * Pp), 256, 0, stream>>>(ef1, F(pb + 12), nullptr, att);
    softmax27<<<blocks(Bz * 4 * Pp), 256, 0, stream>>>(att);
    att_scale<<<blocks((long)64 * Mpp), 256, 0, stream>>>(ef1, att, efs);
    cbl(F(pb + 3), efs, hb, 32, 64, Mpp, F(pb + 4), F(pb + 5));       // hc1
    cbl(F(pb + 6), hb, lgts, 5, 32, Mpp, F(pb + 7), F(pb + 8));       // hc2
    write_h<<<blocks(5 * Mpp), 256, 0, stream>>>(
        lgts, (float*)d_out + (size_t)13 * Mpt + (size_t)hout * 5 * Mpp);
    hop_ghop<<<blocks(Mpp), 256, 0, stream>>>(lgts, ghop);
    cbl(F(pb + 9), efs, ef2, 64, 64, Mpp, F(pb + 10), F(pb + 11));    // ec2
    att_score<<<blocks(Bz * 4 * Pp * Pp), 256, 0, stream>>>(ef2, F(pb + 13), ghop, att);
    softmax27<<<blocks(Bz * 4 * Pp), 256, 0, stream>>>(att);
    att_gmean<<<blocks(64 * Mp), 256, 0, stream>>>(att, ef2, gbuf);
    p2p_add<<<blocks((long)64 * Mpt), 256, 0, stream>>>(gbuf, mask, xi);
  };

  // ---- stage 1: knn(xyz) -> c1,c2 -> max_k -> hop1 ----
  norms_k<<<blocks(Mpt), 256, 0, stream>>>(xc, 6, 3, xx);
  knn_k<<<Mpt, 128, 0, stream>>>(xc, xx, 6, 3, idx);
  cble(F(6), xc, YA, 64, 9, F(7), F(8));                               // c1 (Cin=18)
  cbl(F(9), YA, YB, 64, 64, Mk, F(10), F(11));                         // c2
  maxk_k<<<blocks((long)64 * Mpt), 256, 0, stream>>>(YB, x1);
  hop(31, x1, 0);

  // ---- stage 2 ----
  norms_k<<<blocks(Mpt), 256, 0, stream>>>(x1, 0, 64, xx);
  knn_k<<<Mpt, 128, 0, stream>>>(x1, xx, 0, 64, idx);
  cble(F(12), x1, YA, 64, 64, F(13), F(14));                           // c3 (Cin=128)
  cbl(F(15), YA, YB, 64, 64, Mk, F(16), F(17));                        // c4
  maxk_k<<<blocks((long)64 * Mpt), 256, 0, stream>>>(YB, x2);
  hop(45, x2, 1);

  // ---- stage 3 ----
  norms_k<<<blocks(Mpt), 256, 0, stream>>>(x2, 0, 64, xx);
  knn_k<<<Mpt, 128, 0, stream>>>(x2, xx, 0, 64, idx);
  cble(F(18), x2, YA, 64, 64, F(19), F(20));                           // c5 (Cin=128)
  maxk_k<<<blocks((long)64 * Mpt), 256, 0, stream>>>(YA, x3);
  hop(59, x3, 2);

  // ---- head: c6 -> gmax -> concat -> c7 -> c8 -> c9 ----
  concat3<<<blocks((long)192 * Mpt), 256, 0, stream>>>(x1, x2, x3, cat192);
  cbl(F(21), cat192, YA, 1024, 192, Mpt, F(22), F(23));                // c6
  gmax_k<<<Bz * 1024, 256, 0, stream>>>(YA, gm);
  concat2<<<blocks((long)1216 * Mpt), 256, 0, stream>>>(gm, x1, x2, x3, YB);
  cbl(F(24), YB, o7, 512, 1216, Mpt, F(25), F(26));                    // c7
  cbl(F(27), o7, o8, 256, 512, Mpt, F(28), F(29));                     // c8
  {
    dim3 grid((Mpt + 127) / 128, 1);
    gemm_wmma<<<grid, 256, 0, stream>>>(F(30), o8, o9, 13, 256, Mpt);  // c9 (no BN)
  }
  write_o<<<blocks(13 * Mpt), 256, 0, stream>>>(o9, (float*)d_out);
}
